// QuanvolutionQLSTM_65481071404531
// MI455X (gfx1250) — matrix operand order repaired and checked by reading, not compiled, not run on hardware
//
#include <hip/hip_runtime.h>

// ---------------- problem constants ----------------
#define BATCH   4096
#define HIDDEN  128
#define INDIM   4
#define NCLS    10
#define NP      196          // 14*14 timesteps
#define COMB    132          // INDIM + HIDDEN (reference order: [x, h])
#define KPAD    160          // padded K: [h(128) | x(4) | zeros(28)], 5 k-steps of 32
#define KSTEPS  5

#define WAVES            8                        // one wave per 16-col N-tile
#define ROWS_PER_BLOCK   16                       // batch rows per block
#define NBLOCKS          (BATCH / ROWS_PER_BLOCK) // 256

typedef __attribute__((ext_vector_type(16))) __bf16 v16bf;
typedef __attribute__((ext_vector_type(8)))  float  v8f;

union Frag { v16bf v; uint4 q[2]; };

__device__ __forceinline__ unsigned short f2bf(float f) {
    unsigned u = __float_as_uint(f);
    u += 0x7fffu + ((u >> 16) & 1u);              // round-to-nearest-even
    return (unsigned short)(u >> 16);
}
__device__ __forceinline__ float bf2f(unsigned short h) {
    return __uint_as_float(((unsigned)h) << 16);
}
__device__ __forceinline__ float fsigmoid(float x) {
    return __builtin_amdgcn_rcpf(1.0f + __expf(-x));
}
__device__ __forceinline__ float ftanh(float x) {
    float e = __expf(-2.0f * x);
    return (1.0f - e) * __builtin_amdgcn_rcpf(1.0f + e);
}
__device__ __forceinline__ v8f zero8() {
    v8f z = {0.f, 0.f, 0.f, 0.f, 0.f, 0.f, 0.f, 0.f};
    return z;
}

// ---------------------------------------------------------------------------
// Kernel 1: 2x2/stride-2 conv -> bf16 sequence (NP, BATCH, INDIM)
// ---------------------------------------------------------------------------
__global__ void __launch_bounds__(256)
conv_kernel(const float* __restrict__ x, const float* __restrict__ cw,
            const float* __restrict__ cb, unsigned short* __restrict__ seq) {
    int id = blockIdx.x * 256 + threadIdx.x;      // id = t*BATCH + b
    if (id >= NP * BATCH) return;
    int t = id >> 12;                             // / 4096
    int b = id & (BATCH - 1);
    int py = t / 14, px = t % 14;
    const float* xp = x + (size_t)b * 784 + (2 * py) * 28 + 2 * px;
    float p0 = xp[0], p1 = xp[1], p2 = xp[28], p3 = xp[29];
    unsigned short o[4];
#pragma unroll
    for (int c = 0; c < 4; ++c) {
        float v = cb[c] + p0 * cw[c * 4 + 0] + p1 * cw[c * 4 + 1]
                        + p2 * cw[c * 4 + 2] + p3 * cw[c * 4 + 3];
        o[c] = f2bf(v);
    }
    *(uint2*)(seq + (size_t)id * 4) = *(const uint2*)o;
}

// ---------------------------------------------------------------------------
// Kernel 2: pack [Wh | Wx | 0] per gate as bf16 (4, 128, 160)
// reference comb = [x(0..3), h(4..131)] => col k<128 -> W[j][4+k], 128..131 -> W[j][k-128]
// ---------------------------------------------------------------------------
__global__ void __launch_bounds__(256)
pack_kernel(const float* __restrict__ Wf, const float* __restrict__ Wi,
            const float* __restrict__ Wg, const float* __restrict__ Wo,
            unsigned short* __restrict__ wp) {
    int id = blockIdx.x * 256 + threadIdx.x;      // 4*128*160 = 81920 total
    if (id >= 4 * HIDDEN * KPAD) return;
    int g   = id / (HIDDEN * KPAD);
    int rem = id - g * (HIDDEN * KPAD);
    int j   = rem / KPAD;
    int k   = rem - j * KPAD;
    const float* W = (g == 0) ? Wf : (g == 1) ? Wi : (g == 2) ? Wg : Wo;
    float v = 0.f;
    if (k < HIDDEN)              v = W[j * COMB + INDIM + k];
    else if (k < HIDDEN + INDIM) v = W[j * COMB + (k - HIDDEN)];
    wp[id] = f2bf(v);
}

// ---------------------------------------------------------------------------
// Kernel 3: fused 196-step LSTM scan + classifier + log_softmax
//   - per-wave register-resident B fragments (weights loaded ONCE, 160 VGPRs)
//   - steady state per step: 10 ds_load_b128 (A) -> 20 wmma -> elementwise
//   - ping-pong 10KB LDS h/x buffer, one s_barrier per step
// ---------------------------------------------------------------------------
__global__ void __launch_bounds__(WAVES * 32)
lstm_kernel(const unsigned short* __restrict__ wpack,
            const unsigned short* __restrict__ seq,
            const float* __restrict__ bfv, const float* __restrict__ biv,
            const float* __restrict__ bgv, const float* __restrict__ bov,
            const float* __restrict__ clf_w, const float* __restrict__ clf_b,
            float* __restrict__ out) {
    __shared__ __align__(16) unsigned short hbuf[2][ROWS_PER_BLOCK][KPAD]; // 10240 B
    __shared__ float logit_s[ROWS_PER_BLOCK][NCLS];

    const int tid  = threadIdx.x;
    const int lane = tid & 31;
    const int wave = tid >> 5;                    // == N-tile index nt (0..7)
    const int half = lane >> 4;
    const int l16  = lane & 15;
    const int j    = wave * 16 + l16;             // hidden column owned by this lane
    const int gmbase = blockIdx.x * ROWS_PER_BLOCK;

    // ---- B fragments: 4 gates x 5 k-steps, loaded once from global into VGPRs ----
    Frag Breg[4][KSTEPS];
#pragma unroll
    for (int g = 0; g < 4; ++g) {
        const char* wr = (const char*)wpack + (size_t)(g * HIDDEN + j) * (KPAD * 2);
#pragma unroll
        for (int ks = 0; ks < KSTEPS; ++ks) {     // B per-lane: 16 contiguous K at 16*half
            Breg[g][ks].q[0] = *(const uint4*)(wr + ks * 64 + 32 * half);
            Breg[g][ks].q[1] = *(const uint4*)(wr + ks * 64 + 32 * half + 16);
        }
    }
    // ---- per-lane biases (j is fixed for the whole kernel) ----
    const float bF = bfv[j], bI = biv[j], bG = bgv[j], bO = bov[j];

    // ---- zero both ping-pong buffers (h0 = 0, pad cols = 0), stage x_0 ----
    {
        uint4 z; z.x = z.y = z.z = z.w = 0;
        uint4* hz = (uint4*)hbuf;
        for (int i = tid; i < (int)(sizeof(hbuf) / 16); i += WAVES * 32) hz[i] = z;
    }
    __syncthreads();
    if (wave == 0 && lane < 16) {
        const uint2* sp = (const uint2*)(seq + ((size_t)gmbase + lane) * 4);
        *(uint2*)((char*)hbuf[0][lane] + HIDDEN * 2) = *sp;
    }
    __syncthreads();

    float creg[8];
#pragma unroll
    for (int r = 0; r < 8; ++r) creg[r] = 0.f;

    auto step = [&](int t, unsigned short (*cur)[KPAD], unsigned short (*nxt)[KPAD]) {
        // A fragments: rows of [h|x|0]; per-lane 16 contiguous K at 16*half
        Frag A[KSTEPS];
        const char* arow = (const char*)cur[l16];
#pragma unroll
        for (int ks = 0; ks < KSTEPS; ++ks) {
            A[ks].q[0] = *(const uint4*)(arow + ks * 64 + 32 * half);
            A[ks].q[1] = *(const uint4*)(arow + ks * 64 + 32 * half + 16);
        }
        v8f accf = zero8(), acci = zero8(), accg = zero8(), acco = zero8();
#pragma unroll
        for (int ks = 0; ks < KSTEPS; ++ks) {
            accf = __builtin_amdgcn_wmma_f32_16x16x32_bf16(false, A[ks].v, false, Breg[0][ks].v, (short)0, accf, false, false);
            acci = __builtin_amdgcn_wmma_f32_16x16x32_bf16(false, A[ks].v, false, Breg[1][ks].v, (short)0, acci, false, false);
            accg = __builtin_amdgcn_wmma_f32_16x16x32_bf16(false, A[ks].v, false, Breg[2][ks].v, (short)0, accg, false, false);
            acco = __builtin_amdgcn_wmma_f32_16x16x32_bf16(false, A[ks].v, false, Breg[3][ks].v, (short)0, acco, false, false);
        }
        char* nb = (char*)nxt;
#pragma unroll
        for (int r = 0; r < 8; ++r) {             // C/D: row = r + 8*half, col = j
            float f  = fsigmoid(accf[r] + bF);
            float ii = fsigmoid(acci[r] + bI);
            float g  = ftanh   (accg[r] + bG);
            float o  = fsigmoid(acco[r] + bO);
            float c  = f * creg[r] + ii * g;
            creg[r] = c;
            float h  = o * ftanh(c);
            *(unsigned short*)(nb + (r + 8 * half) * (KPAD * 2) + j * 2) = f2bf(h);
        }
        // wave 0 stages x_{t+1}; prefetch x_{t+3} toward the WGP caches
        if (wave == 0 && lane < 16) {
            if (t + 1 < NP) {
                const uint2* sp = (const uint2*)(seq + ((size_t)(t + 1) * BATCH + gmbase + lane) * 4);
                *(uint2*)(nb + lane * (KPAD * 2) + HIDDEN * 2) = *sp;
            }
            if (t + 3 < NP)
                __builtin_prefetch((const void*)(seq + ((size_t)(t + 3) * BATCH + gmbase + lane) * 4), 0, 1);
        }
        __syncthreads();
    };

    for (int t = 0; t < NP; t += 2) {             // NP is even; ping-pong unrolled x2
        step(t,     hbuf[0], hbuf[1]);
        step(t + 1, hbuf[1], hbuf[0]);
    }
    // final h (t = NP-1) landed in hbuf[0]

    // ---- classifier: 160 threads each compute one (row, class) dot ----
    if (tid < ROWS_PER_BLOCK * NCLS) {
        int row = tid / NCLS, cls = tid - row * NCLS;
        const unsigned short* hrow = hbuf[0][row];
        float s = clf_b[cls];
#pragma unroll 8
        for (int jj = 0; jj < HIDDEN; ++jj)
            s += bf2f(hrow[jj]) * clf_w[cls * HIDDEN + jj];
        logit_s[row][cls] = s;
    }
    __syncthreads();
    // ---- log_softmax: one thread per row ----
    if (tid < ROWS_PER_BLOCK) {
        float lg[NCLS];
#pragma unroll
        for (int c = 0; c < NCLS; ++c) lg[c] = logit_s[tid][c];
        float mx = lg[0];
#pragma unroll
        for (int c = 1; c < NCLS; ++c) mx = fmaxf(mx, lg[c]);
        float ss = 0.f;
#pragma unroll
        for (int c = 0; c < NCLS; ++c) ss += __expf(lg[c] - mx);
        float lse = mx + __logf(ss);
        float* op = out + (size_t)(gmbase + tid) * NCLS;
#pragma unroll
        for (int c = 0; c < NCLS; ++c) op[c] = lg[c] - lse;
    }
}

// ---------------------------------------------------------------------------
extern "C" void kernel_launch(void* const* d_in, const int* in_sizes, int n_in,
                              void* d_out, int out_size, void* d_ws, size_t ws_size,
                              hipStream_t stream) {
    (void)in_sizes; (void)n_in; (void)out_size; (void)ws_size;
    const float* x      = (const float*)d_in[0];
    const float* conv_w = (const float*)d_in[1];
    const float* conv_b = (const float*)d_in[2];
    const float* Wf = (const float*)d_in[3];  const float* bfp = (const float*)d_in[4];
    const float* Wi = (const float*)d_in[5];  const float* bip = (const float*)d_in[6];
    const float* Wg = (const float*)d_in[7];  const float* bgp = (const float*)d_in[8];
    const float* Wo = (const float*)d_in[9];  const float* bop = (const float*)d_in[10];
    const float* clf_w = (const float*)d_in[11];
    const float* clf_b = (const float*)d_in[12];

    unsigned short* wpack = (unsigned short*)d_ws;                               // 160 KB
    unsigned short* seq   = (unsigned short*)((char*)d_ws + 4 * HIDDEN * KPAD * 2); // 6.4 MB

    pack_kernel<<<(4 * HIDDEN * KPAD + 255) / 256, 256, 0, stream>>>(Wf, Wi, Wg, Wo, wpack);
    conv_kernel<<<(NP * BATCH + 255) / 256, 256, 0, stream>>>(x, conv_w, conv_b, seq);

    lstm_kernel<<<NBLOCKS, WAVES * 32, 0, stream>>>(
        wpack, seq, bfp, bip, bgp, bop, clf_w, clf_b, (float*)d_out);
}
